// Block_16106127360642
// MI455X (gfx1250) — matrix operand order repaired and checked by reading, not compiled, not run on hardware
//
#include <hip/hip_runtime.h>
#include <hip/hip_bf16.h>

// ---------------------------------------------------------------------------
// Types for CDNA5 WMMA (wave32): v16bf A/B fragments, v8f C/D fragments.
// ---------------------------------------------------------------------------
typedef __attribute__((ext_vector_type(16))) __bf16         v16bf;
typedef __attribute__((ext_vector_type(16))) unsigned short v16us;
typedef __attribute__((ext_vector_type(8)))  float          v8f;

union BF16Frag { v16us u; v16bf b; };
union U4x8     { uint4 v; unsigned short s[8]; };

__device__ __forceinline__ unsigned short f2bf(float f) {
  union { float f; unsigned int u; } x; x.f = f;
  unsigned int r = x.u + 0x7FFFu + ((x.u >> 16) & 1u);  // round-nearest-even
  return (unsigned short)(r >> 16);
}

__device__ __forceinline__ v8f v8f_zero() {
  v8f z;
#pragma unroll
  for (int i = 0; i < 8; ++i) z[i] = 0.0f;
  return z;
}

#define WMMA_BF16(A, B, Cacc) \
  __builtin_amdgcn_wmma_f32_16x16x32_bf16(false, (A), false, (B), (short)0, (Cacc), false, false)

// CDNA5 async global->LDS DMA (GVS mode: SGPR base + per-lane u32 byte offset;
// vdst VGPR carries the per-lane LDS byte address). Tracked by ASYNCcnt.
#define ASYNC_G2L_B128(ldsOff, gByteOff, sBase)                               \
  asm volatile("global_load_async_to_lds_b128 %0, %1, %2"                     \
               :: "v"(ldsOff), "v"(gByteOff), "s"(sBase) : "memory")
#define WAIT_ASYNC0() asm volatile("s_wait_asynccnt 0x0" ::: "memory")

__device__ __forceinline__ unsigned lds_off(const void* p) {
  return (unsigned)(uintptr_t)p;  // generic shared ptr: low 32 bits = LDS offset
}

// ---------------------------------------------------------------------------
// FP32 -> BF16 elementwise convert (weights, once per call).
// ---------------------------------------------------------------------------
__global__ void cvt_f32_to_bf16(const float* __restrict__ in,
                                unsigned short* __restrict__ out, int n) {
  int i = blockIdx.x * blockDim.x + threadIdx.x;
  if (i < n) out[i] = f2bf(in[i]);
}

// ---------------------------------------------------------------------------
// LayerNorm over last dim (C=768), one block per row, output BF16.
// ---------------------------------------------------------------------------
__global__ __launch_bounds__(256) void layernorm_bf16(
    const float* __restrict__ x, const float* __restrict__ g,
    const float* __restrict__ b, unsigned short* __restrict__ out, int C) {
  const int row = blockIdx.x;
  const float* xr = x + (long)row * C;
  __shared__ float red[8];
  const int wave = threadIdx.x >> 5;
  const int nw = blockDim.x >> 5;

  float s = 0.f;
  for (int c = threadIdx.x; c < C; c += blockDim.x) s += xr[c];
#pragma unroll
  for (int m = 16; m >= 1; m >>= 1) s += __shfl_xor(s, m, 32);
  if ((threadIdx.x & 31) == 0) red[wave] = s;
  __syncthreads();
  float mu = 0.f;
  for (int w = 0; w < nw; ++w) mu += red[w];
  mu /= (float)C;
  __syncthreads();

  float v = 0.f;
  for (int c = threadIdx.x; c < C; c += blockDim.x) {
    float d = xr[c] - mu; v += d * d;
  }
#pragma unroll
  for (int m = 16; m >= 1; m >>= 1) v += __shfl_xor(v, m, 32);
  if ((threadIdx.x & 31) == 0) red[wave] = v;
  __syncthreads();
  float var = 0.f;
  for (int w = 0; w < nw; ++w) var += red[w];
  var /= (float)C;
  const float rs = rsqrtf(var + 1e-5f);

  for (int c = threadIdx.x; c < C; c += blockDim.x)
    out[(long)row * C + c] = f2bf((xr[c] - mu) * rs * g[c] + b[c]);
}

// ---------------------------------------------------------------------------
// Tiled BF16 WMMA GEMM: C[M,N] = A[M,K] x B[K,N]  (A,B bf16 row-major).
// Block tile 128x128, 256 threads = 8 waves (2x4), wave tile 64x32 (4x2 WMMA).
// A tile staged with CDNA5 async global->LDS DMA; B tile staged manually
// (transposed into LDS so fragment K-pairs coalesce into ds_load_b128).
// EPI: 0 = store bf16; 1 = +bias +residual -> f32; 2 = +bias, GELU -> bf16.
// ---------------------------------------------------------------------------
#define BM 128
#define BN 128
#define BK 32

template <int EPI>
__global__ __launch_bounds__(256) void gemm_bf16_wmma(
    const unsigned short* __restrict__ A, const unsigned short* __restrict__ Bm,
    const float* __restrict__ bias, const float* __restrict__ residual,
    void* __restrict__ out, int M, int N, int K) {
  __shared__ unsigned short As[BM][BK];   // 8 KB
  __shared__ unsigned short BsT[BN][BK];  // 8 KB (transposed: [n][k])

  const int tid  = threadIdx.x;
  const int lane = tid & 31;
  const int wave = tid >> 5;
  const int wm   = wave >> 2;  // 0..1
  const int wn   = wave & 3;   // 0..3
  const int blockM = blockIdx.y * BM;
  const int blockN = blockIdx.x * BN;

  const int mhalf = lane & 15;         // A fragment row
  const int khalf = (lane >> 4) * 8;   // A fragment K sub-offset
  const int koffB = (lane >> 4) * 16;  // unused (B fragment also uses khalf)
  const int nlane = lane & 15;         // B fragment column

  v8f acc[4][2];
#pragma unroll
  for (int i = 0; i < 4; ++i)
#pragma unroll
    for (int j = 0; j < 2; ++j) acc[i][j] = v8f_zero();

  for (int k0 = 0; k0 < K; k0 += BK) {
    // ---- stage A tile (128x32) via async DMA: 512 x b128, 2 per thread --
#pragma unroll
    for (int it = 0; it < 2; ++it) {
      int u = tid + it * 256;         // 0..511
      int r = u >> 2, seg = u & 3;    // 4 x b128 per 32-elem row
      unsigned gOff = (unsigned)(((long)(blockM + r) * K + k0 + seg * 8) * 2);
      ASYNC_G2L_B128(lds_off(&As[r][seg * 8]), gOff, A);
    }
    // ---- stage B tile (32x128) transposed into BsT[n][k] ----------------
#pragma unroll
    for (int it = 0; it < 2; ++it) {
      int u = tid + it * 256;         // 0..511
      int r = u >> 4, seg = u & 15;   // r = k row, seg*8 = n base
      U4x8 val;
      val.v = *(const uint4*)(Bm + (long)(k0 + r) * N + blockN + seg * 8);
#pragma unroll
      for (int e = 0; e < 8; ++e) BsT[seg * 8 + e][r] = val.s[e];
    }
    WAIT_ASYNC0();
    __syncthreads();

    if (k0 + BK < K) {  // gfx1250 global_prefetch_b8 on next B tile
      __builtin_prefetch(Bm + (long)(k0 + BK + (tid & 31)) * N + blockN, 0, 1);
    }

    // ---- load fragments --------------------------------------------------
    BF16Frag af[4], bfr[2];
#pragma unroll
    for (int i = 0; i < 4; ++i) {
      const int m = wm * 64 + i * 16 + mhalf;
      v16us t;
#pragma unroll
      for (int v = 0; v < 8; ++v) {
        int kp = ((v < 4) ? 2 * v : 16 + 2 * (v - 4)) + khalf;
        t[2 * v]     = As[m][kp];
        t[2 * v + 1] = As[m][kp + 1];
      }
      af[i].u = t;
    }
#pragma unroll
    for (int j = 0; j < 2; ++j) {
      const int n = wn * 32 + j * 16 + nlane;
      v16us t;
#pragma unroll
      for (int v = 0; v < 8; ++v) {
        int kp = ((v < 4) ? 2 * v : 16 + 2 * (v - 4)) + khalf;
        t[2 * v]     = BsT[n][kp];
        t[2 * v + 1] = BsT[n][kp + 1];
      }
      bfr[j].u = t;
    }

    // ---- 8 WMMAs ---------------------------------------------------------
#pragma unroll
    for (int i = 0; i < 4; ++i)
#pragma unroll
      for (int j = 0; j < 2; ++j)
        acc[i][j] = WMMA_BF16(af[i].b, bfr[j].b, acc[i][j]);

    __syncthreads();
  }

  // ---- epilogue: C layout row = r + 8*(lane>>4), col = lane&15 ----------
  const int rowb = (lane >> 4) * 8;
  const int colb = lane & 15;
#pragma unroll
  for (int i = 0; i < 4; ++i) {
#pragma unroll
    for (int j = 0; j < 2; ++j) {
      const int gn = blockN + wn * 32 + j * 16 + colb;
#pragma unroll
      for (int r = 0; r < 8; ++r) {
        const int gm = blockM + wm * 64 + i * 16 + rowb + r;
        float v = acc[i][j][r];
        if (EPI == 0) {
          ((unsigned short*)out)[(long)gm * N + gn] = f2bf(v);
        } else if (EPI == 1) {
          v += bias[gn] + residual[(long)gm * N + gn];
          ((float*)out)[(long)gm * N + gn] = v;
        } else {
          v += bias[gn];
          v = 0.5f * v * (1.0f + erff(v * 0.70710678118654752f));  // exact GELU
          ((unsigned short*)out)[(long)gm * N + gn] = f2bf(v);
        }
      }
    }
  }
}

// ---------------------------------------------------------------------------
// Flash attention, BF16 WMMA. qkv: [Nseq, 3*C] (col = which*C + h*64 + d).
// Block = 4 waves; wave handles 16 query rows of one head. Key tiles of 32.
// K tile staged with async global->LDS DMA; V tile transposed manually.
// ---------------------------------------------------------------------------
__global__ __launch_bounds__(128) void flash_attn_bf16(
    const unsigned short* __restrict__ qkv, unsigned short* __restrict__ out,
    int Nseq) {
  const int C = 768, Dh = 64, W3 = 3 * C;
  const int h    = blockIdx.y;
  const int tid  = threadIdx.x;
  const int lane = tid & 31;
  const int wave = tid >> 5;  // 0..3
  const int qBase = blockIdx.x * 64 + wave * 16;

  __shared__ unsigned short Ks[32][64];      // K tile, row-major [key][d]
  __shared__ unsigned short Vt[64][32];      // V tile transposed [d][key]
  __shared__ unsigned short Ps[4][16][32];   // per-wave P scratch

  const int mhalf = lane & 15;
  const int khalf = (lane >> 4) * 8;
  const int koffB = (lane >> 4) * 16;
  const int nlane = lane & 15;
  const float scale = 0.125f;  // 1/sqrt(64)

  // Q fragments (A-layout), two K=32 chunks covering Dh=64
  BF16Frag qf[2];
#pragma unroll
  for (int c = 0; c < 2; ++c) {
    const unsigned short* qrow = qkv + (long)(qBase + mhalf) * W3 + h * Dh + c * 32;
    v16us t;
#pragma unroll
    for (int v = 0; v < 8; ++v) {
      int kp = ((v < 4) ? 2 * v : 16 + 2 * (v - 4)) + khalf;
      t[2 * v]     = qrow[kp];
      t[2 * v + 1] = qrow[kp + 1];
    }
    qf[c].u = t;
  }

  v8f o[4];
#pragma unroll
  for (int n = 0; n < 4; ++n) o[n] = v8f_zero();
  float mrow[8], lrow[8];
#pragma unroll
  for (int r = 0; r < 8; ++r) { mrow[r] = -3.0e38f; lrow[r] = 0.0f; }

  for (int kt = 0; kt < Nseq; kt += 32) {
    // ---- stage K tile [32][64] via async DMA: 256 x b128, 2 per thread --
#pragma unroll
    for (int it = 0; it < 2; ++it) {
      int u = tid + it * 128;
      int r = u >> 3, seg = u & 7;
      unsigned gOff = (unsigned)(((long)(kt + r) * W3 + C + h * Dh + seg * 8) * 2);
      ASYNC_G2L_B128(lds_off(&Ks[r][seg * 8]), gOff, qkv);
    }
    // ---- stage V tile transposed: thread reads 16 contiguous d ----------
    {
      int r = tid >> 2;                 // key 0..31
      int d0 = (tid & 3) * 16;          // d base
      const unsigned short* vrow = qkv + (long)(kt + r) * W3 + 2 * C + h * Dh + d0;
#pragma unroll
      for (int d = 0; d < 16; ++d) Vt[d0 + d][r] = vrow[d];
    }
    WAIT_ASYNC0();
    __syncthreads();

    // ---- S = Q * K^T : two 16x16 tiles over the 32 keys -----------------
    v8f s[2];
#pragma unroll
    for (int j = 0; j < 2; ++j) {
      v8f sa = v8f_zero();
#pragma unroll
      for (int c = 0; c < 2; ++c) {
        BF16Frag kb;
        const int key = j * 16 + nlane;
        v16us t;
#pragma unroll
        for (int v = 0; v < 8; ++v) {
          int d = c * 32 + ((v < 4) ? 2 * v : 16 + 2 * (v - 4)) + khalf;
          t[2 * v]     = Ks[key][d];
          t[2 * v + 1] = Ks[key][d + 1];
        }
        kb.u = t;
        sa = WMMA_BF16(qf[c].b, kb.b, sa);
      }
      s[j] = sa;
    }

    // ---- online softmax (rows live across 16 lanes of each half) --------
    float p0r[8], p1r[8], alpha[8];
#pragma unroll
    for (int r = 0; r < 8; ++r) {
      float s0 = s[0][r] * scale, s1 = s[1][r] * scale;
      float mx = fmaxf(s0, s1);
#pragma unroll
      for (int msk = 8; msk >= 1; msk >>= 1) mx = fmaxf(mx, __shfl_xor(mx, msk, 32));
      float mnew = fmaxf(mrow[r], mx);
      alpha[r] = __expf(mrow[r] - mnew);
      float p0 = __expf(s0 - mnew), p1 = __expf(s1 - mnew);
      float ps = p0 + p1;
#pragma unroll
      for (int msk = 8; msk >= 1; msk >>= 1) ps += __shfl_xor(ps, msk, 32);
      lrow[r] = lrow[r] * alpha[r] + ps;
      mrow[r] = mnew;
      p0r[r] = p0; p1r[r] = p1;
    }

    // ---- P -> per-wave LDS (bf16), cross-lane shuffle to A-layout -------
    {
      const int rowb = (lane >> 4) * 8, colb = lane & 15;
#pragma unroll
      for (int r = 0; r < 8; ++r) {
        Ps[wave][rowb + r][colb]      = f2bf(p0r[r]);
        Ps[wave][rowb + r][16 + colb] = f2bf(p1r[r]);
      }
    }
    // DS ops are in-order per wave; enforce ordering + drain before reread.
    asm volatile("s_wait_dscnt 0x0" ::: "memory");

#pragma unroll
    for (int n = 0; n < 4; ++n)
#pragma unroll
      for (int r = 0; r < 8; ++r) o[n][r] *= alpha[r];

    BF16Frag pf;
    {
      v16us t;
#pragma unroll
      for (int v = 0; v < 8; ++v) {
        int kp = ((v < 4) ? 2 * v : 16 + 2 * (v - 4)) + khalf;
        t[2 * v]     = Ps[wave][mhalf][kp];
        t[2 * v + 1] = Ps[wave][mhalf][kp + 1];
      }
      pf.u = t;
    }

    // ---- O += P * V (contraction over 32 keys; 4 d-subtiles) ------------
#pragma unroll
    for (int n = 0; n < 4; ++n) {
      BF16Frag vb;
      const int d = n * 16 + nlane;
      v16us t;
#pragma unroll
      for (int v = 0; v < 8; ++v) {
        int key = ((v < 4) ? 2 * v : 16 + 2 * (v - 4)) + khalf;
        t[2 * v]     = Vt[d][key];
        t[2 * v + 1] = Vt[d][key + 1];
      }
      vb.u = t;
      o[n] = WMMA_BF16(pf.b, vb.b, o[n]);
    }
    __syncthreads();
  }

  // ---- finalize: O /= l, store bf16 to [n, h*64 + d] --------------------
  const int rowb = (lane >> 4) * 8, colb = lane & 15;
#pragma unroll
  for (int n = 0; n < 4; ++n)
#pragma unroll
    for (int r = 0; r < 8; ++r) {
      float v = o[n][r] / lrow[r];
      out[(long)(qBase + rowb + r) * C + h * Dh + n * 16 + colb] = f2bf(v);
    }
}

// ---------------------------------------------------------------------------
// Launch: LN1 -> QKV gemm -> attention -> proj(+res) -> LN2 -> FC1(GELU)
//         -> FC2(+res) = output.
// ---------------------------------------------------------------------------
extern "C" void kernel_launch(void* const* d_in, const int* in_sizes, int n_in,
                              void* d_out, int out_size, void* d_ws, size_t ws_size,
                              hipStream_t stream) {
  (void)in_sizes; (void)n_in; (void)out_size; (void)ws_size;
  const float* x      = (const float*)d_in[0];
  const float* w_qkv  = (const float*)d_in[1];
  const float* w_proj = (const float*)d_in[2];
  const float* b_proj = (const float*)d_in[3];
  const float* g1     = (const float*)d_in[4];
  const float* b1     = (const float*)d_in[5];
  const float* g2     = (const float*)d_in[6];
  const float* b2     = (const float*)d_in[7];
  const float* w_fc1  = (const float*)d_in[8];
  const float* b_fc1  = (const float*)d_in[9];
  const float* w_fc2  = (const float*)d_in[10];
  const float* b_fc2  = (const float*)d_in[11];

  const int Nseq = 4096, C = 768, C3 = 3 * C, HID = 3072, H = 12;

  char* ws = (char*)d_ws;
  size_t off = 0;
  auto alloc = [&](size_t bytes) -> void* {
    void* p = ws + off;
    off += (bytes + 255) & ~(size_t)255;
    return p;
  };
  unsigned short* wqkv_bf  = (unsigned short*)alloc((size_t)C * C3 * 2);
  unsigned short* wproj_bf = (unsigned short*)alloc((size_t)C * C * 2);
  unsigned short* wfc1_bf  = (unsigned short*)alloc((size_t)C * HID * 2);
  unsigned short* wfc2_bf  = (unsigned short*)alloc((size_t)HID * C * 2);
  unsigned short* h1_bf    = (unsigned short*)alloc((size_t)Nseq * C * 2);
  unsigned short* qkv_bf   = (unsigned short*)alloc((size_t)Nseq * C3 * 2);
  unsigned short* attn_bf  = (unsigned short*)alloc((size_t)Nseq * C * 2);
  float*          x1_f32   = (float*)alloc((size_t)Nseq * C * 4);
  unsigned short* h2_bf    = (unsigned short*)alloc((size_t)Nseq * C * 2);
  unsigned short* hfc_bf   = (unsigned short*)alloc((size_t)Nseq * HID * 2);

  // Weight conversion (deterministic every call)
  cvt_f32_to_bf16<<<(C * C3 + 255) / 256, 256, 0, stream>>>(w_qkv, wqkv_bf, C * C3);
  cvt_f32_to_bf16<<<(C * C + 255) / 256, 256, 0, stream>>>(w_proj, wproj_bf, C * C);
  cvt_f32_to_bf16<<<(C * HID + 255) / 256, 256, 0, stream>>>(w_fc1, wfc1_bf, C * HID);
  cvt_f32_to_bf16<<<(HID * C + 255) / 256, 256, 0, stream>>>(w_fc2, wfc2_bf, HID * C);

  // LN1
  layernorm_bf16<<<Nseq, 256, 0, stream>>>(x, g1, b1, h1_bf, C);

  // QKV GEMM: [4096,768] x [768,2304] -> bf16 [4096,2304]
  gemm_bf16_wmma<0><<<dim3(C3 / BN, Nseq / BM), 256, 0, stream>>>(
      h1_bf, wqkv_bf, nullptr, nullptr, qkv_bf, Nseq, C3, C);

  // Attention: 64 query tiles x 12 heads
  flash_attn_bf16<<<dim3(Nseq / 64, H), 128, 0, stream>>>(qkv_bf, attn_bf, Nseq);

  // Proj + bias + residual(x) -> fp32 x1
  gemm_bf16_wmma<1><<<dim3(C / BN, Nseq / BM), 256, 0, stream>>>(
      attn_bf, wproj_bf, b_proj, x, x1_f32, Nseq, C, C);

  // LN2
  layernorm_bf16<<<Nseq, 256, 0, stream>>>(x1_f32, g2, b2, h2_bf, C);

  // FC1 + bias + GELU -> bf16
  gemm_bf16_wmma<2><<<dim3(HID / BN, Nseq / BM), 256, 0, stream>>>(
      h2_bf, wfc1_bf, b_fc1, nullptr, hfc_bf, Nseq, HID, C);

  // FC2 + bias + residual(x1) -> fp32 output
  gemm_bf16_wmma<1><<<dim3(C / BN, Nseq / BM), 256, 0, stream>>>(
      hfc_bf, wfc2_bf, b_fc2, x1_f32, (float*)d_out, Nseq, C, HID);
}
